// Attention_59785944760577
// MI455X (gfx1250) — matrix operand order repaired and checked by reading, not compiled, not run on hardware
//
#include <hip/hip_runtime.h>

// ---------------------------------------------------------------------------
// Shapes: B=8, T=4096, D=512.  out = causal softmax-pooled context [B,T,D] f32
// Pipeline:
//   k1: W1 f32 -> W1t bf16 transposed [n][k]      (1 MB read, trivial)
//   k2: s[b,t] = w2 . tanh(x@W1 + b1) + b2        (bf16 WMMA, TDM loads x)
//   k3: m[b] = rowmax(s)
//   k4: e = exp(s-m); per-256-token-chunk sums of e and e*x
//   k5: exclusive prefix of chunk sums (tiny)
//   k6: per-chunk sequential scan -> out          (bandwidth bound, 128 WGs)
// ---------------------------------------------------------------------------

typedef __attribute__((ext_vector_type(4)))  unsigned int u32x4;
typedef __attribute__((ext_vector_type(8)))  unsigned int u32x8;
typedef __attribute__((ext_vector_type(8)))  int          i32x8;
typedef __attribute__((ext_vector_type(4)))  int          i32x4;
typedef __attribute__((ext_vector_type(8)))  float        v8f;
typedef __attribute__((ext_vector_type(16))) __bf16       v16bf;

#define D_DIM   512
#define T_DIM   4096
#define B_DIM   8
#define NTOK    (B_DIM * T_DIM)     // 32768
#define CHUNK   256
#define NCHUNK  (NTOK / CHUNK)      // 128 total (16 per batch row)

__device__ __forceinline__ unsigned int pk_bf16(float a, float b) {
    unsigned int ua = __builtin_bit_cast(unsigned int, a);
    unsigned int ub = __builtin_bit_cast(unsigned int, b);
    ua += 0x7FFFu + ((ua >> 16) & 1u);   // round-to-nearest-even (sans NaN care)
    ub += 0x7FFFu + ((ub >> 16) & 1u);
    return (ua >> 16) | (ub & 0xFFFF0000u);
}

// ---------------- k1: W1[k][n] f32 -> W1t[n][k] bf16 ------------------------
__global__ void cvt_w1_kernel(const float* __restrict__ w1,
                              unsigned short* __restrict__ w1t) {
    int gid  = blockIdx.x * 256 + threadIdx.x;   // 131072 threads, 2 elems each
    int base = gid * 2;
    int n = base >> 9;
    int k = base & 511;
    float a = w1[(long long)k * D_DIM + n];
    float b = w1[(long long)(k + 1) * D_DIM + n];
    *(unsigned int*)(w1t + (long long)n * D_DIM + k) = pk_bf16(a, b);
}

// ---------------- k2: scores via bf16 WMMA + TDM ----------------------------
// grid 512 x 128 threads (4 waves). wave -> 16-token tile.
// LDS per wave (32 KB): [0,16K) fp32 staging (8x512, TDM dest)
//                       [16K,32K) bf16 tile  (16x512 row-major)
__global__ __launch_bounds__(128, 1)
void scores_kernel(const float* __restrict__ x,
                   const unsigned short* __restrict__ w1t,
                   const float* __restrict__ b1,
                   const float* __restrict__ w2,
                   const float* __restrict__ b2,
                   float* __restrict__ s_buf) {
    extern __shared__ char smem[];
    const int lane = threadIdx.x & 31;
    const int wave = threadIdx.x >> 5;
    const int tile = blockIdx.x * 4 + wave;
    const long long r0 = (long long)tile * 16;           // first token row

    float*          stg = (float*)(smem + wave * 32768);
    unsigned short* bfs = (unsigned short*)(smem + wave * 32768 + 16384);
    const unsigned  lds_base = (unsigned)(unsigned long long)smem + wave * 32768u;

    // --- stage x tile: 2 rounds of (TDM 8x512 f32 -> LDS) + convert to bf16
    for (int part = 0; part < 2; ++part) {
        unsigned long long ga =
            (unsigned long long)(const void*)(x + (r0 + part * 8) * D_DIM);
        u32x4 g0 = (u32x4)(0u);
        g0[0] = 1u;                                       // count=1, user mode
        g0[1] = lds_base;                                 // lds_addr
        g0[2] = (unsigned)(ga & 0xFFFFFFFFull);           // global_addr[31:0]
        g0[3] = (unsigned)((ga >> 32) & 0x01FFFFFFull) |  // global_addr[56:32]
                (2u << 30);                               // type = 2 (image)
        i32x8 g1 = (i32x8)(0);
        g1[0] = (int)(2u << 16);                          // data_size = 4 bytes
        g1[1] = (int)(512u << 16);                        // tensor_dim0 = 512
        g1[2] = (int)(8u << 16);                          // tensor_dim1 = 8
        g1[3] = (int)(512u << 16);                        // tile_dim0 = 512
        g1[4] = (int)(8u);                                // tile_dim1 = 8
        g1[5] = (int)(512u);                              // tensor_dim0_stride
        i32x4 g2 = (i32x4)(0);
        i32x4 g3 = (i32x4)(0);
        i32x8 g4 = (i32x8)(0);                            // 6-arg form extra group
        __builtin_amdgcn_tensor_load_to_lds(g0, g1, g2, g3, g4, 0);
        __builtin_amdgcn_s_wait_tensorcnt(0);

        // convert this 8x512 slab to bf16 (each lane: 128 floats)
        for (int i = 0; i < 32; ++i) {
            int idx = i * 128 + lane * 4;
            float4 f = *(const float4*)(stg + idx);
            unsigned lo = pk_bf16(f.x, f.y);
            unsigned hi = pk_bf16(f.z, f.w);
            *(uint2*)(bfs + part * 4096 + idx) = make_uint2(lo, hi);
        }
    }

    // --- WMMA: 16x512 (A, bf16 from LDS) x 512x512 (B = W1t from L2)
    const int m    = lane & 15;
    const int half = lane >> 4;
    float sp[8] = {};                                     // per-lane s partials

    for (int ngrp = 0; ngrp < 4; ++ngrp) {                // 8 n-tiles / group
        v8f acc[8] = {};
        for (int ks = 0; ks < 16; ++ks) {
            const int kb = ks * 32;
            // A fragment: lanes 0-15 -> K kb+0..7 | kb+16..23 ; lanes 16-31 +8
            u32x4 a0 = *(const u32x4*)(bfs + m * D_DIM + kb + half * 8);
            u32x4 a1 = *(const u32x4*)(bfs + m * D_DIM + kb + 16 + half * 8);
            u32x8 au;
            au[0] = a0[0]; au[1] = a0[1]; au[2] = a0[2]; au[3] = a0[3];
            au[4] = a1[0]; au[5] = a1[1]; au[6] = a1[2]; au[7] = a1[3];
            v16bf a = __builtin_bit_cast(v16bf, au);
            const int krow = kb + half * 8;
#pragma unroll
            for (int j = 0; j < 8; ++j) {
                const int n = (ngrp * 8 + j) * 16 + m;    // B column
                u32x4 b0 = *(const u32x4*)(w1t + (long long)n * D_DIM + krow);
                u32x4 b1v4 = *(const u32x4*)(w1t + (long long)n * D_DIM + krow + 16);
                u32x8 bu;
                bu[0] = b0[0]; bu[1] = b0[1]; bu[2] = b0[2]; bu[3] = b0[3];
                bu[4] = b1v4[0]; bu[5] = b1v4[1]; bu[6] = b1v4[2]; bu[7] = b1v4[3];
                v16bf b = __builtin_bit_cast(v16bf, bu);
                acc[j] = __builtin_amdgcn_wmma_f32_16x16x32_bf16(
                    false, a, false, b, (short)0, acc[j], false, false);
            }
        }
        // epilogue: sp[i] += w2[n] * tanh(acc + b1[n])
#pragma unroll
        for (int j = 0; j < 8; ++j) {
            const int n  = (ngrp * 8 + j) * 16 + m;
            const float w2v = w2[n];
            const float b1v = b1[n];
#pragma unroll
            for (int i = 0; i < 8; ++i)
                sp[i] += w2v * tanhf(acc[j][i] + b1v);
        }
    }
    // reduce over N: butterfly within each 16-lane half (wave32)
#pragma unroll
    for (int i = 0; i < 8; ++i) {
        float v = sp[i];
        v += __shfl_xor(v, 1);
        v += __shfl_xor(v, 2);
        v += __shfl_xor(v, 4);
        v += __shfl_xor(v, 8);
        sp[i] = v;
    }
    if (m == 0) {                                         // lanes 0 and 16
        const float b2s = b2[0];
#pragma unroll
        for (int i = 0; i < 8; ++i)
            s_buf[r0 + half * 8 + i] = sp[i] + b2s;       // rows M=0..7 / 8..15
    }
}

// ---------------- k3: per-batch row max -------------------------------------
__global__ void rowmax_kernel(const float* __restrict__ s,
                              float* __restrict__ m) {
    __shared__ float red[256];
    const int b = blockIdx.x;
    float mx = -3.402823466e+38f;
    for (int t = threadIdx.x; t < T_DIM; t += 256)
        mx = fmaxf(mx, s[b * T_DIM + t]);
    red[threadIdx.x] = mx;
    __syncthreads();
    for (int off = 128; off > 0; off >>= 1) {
        if (threadIdx.x < off)
            red[threadIdx.x] = fmaxf(red[threadIdx.x], red[threadIdx.x + off]);
        __syncthreads();
    }
    if (threadIdx.x == 0) m[b] = red[0];
}

// ---------------- k4: e = exp(s-m); chunk sums of e and e*x -----------------
__global__ __launch_bounds__(256)
void chunksum_kernel(const float* __restrict__ x,
                     const float* __restrict__ s,
                     const float* __restrict__ m,
                     float* __restrict__ e_buf,
                     float* __restrict__ cse,
                     float* __restrict__ csx) {
    __shared__ float es[CHUNK];
    __shared__ float red[256];
    const int b = blockIdx.x >> 4;
    const int c = blockIdx.x & 15;
    const int t = c * CHUNK + (int)threadIdx.x;
    const float e = expf(s[b * T_DIM + t] - m[b]);
    es[threadIdx.x] = e;
    e_buf[b * T_DIM + t] = e;
    red[threadIdx.x] = e;
    __syncthreads();
    for (int off = 128; off > 0; off >>= 1) {
        if (threadIdx.x < off) red[threadIdx.x] += red[threadIdx.x + off];
        __syncthreads();
    }
    if (threadIdx.x == 0) cse[blockIdx.x] = red[0];
    // sums of e*x over the chunk, per channel d (coalesced across threads)
    for (int d = threadIdx.x; d < D_DIM; d += 256) {
        const float* xp = x + ((long long)b * T_DIM + c * CHUNK) * D_DIM + d;
        float acc = 0.f;
        for (int tt = 0; tt < CHUNK; ++tt)
            acc = fmaf(es[tt], xp[(long long)tt * D_DIM], acc);
        csx[(long long)blockIdx.x * D_DIM + d] = acc;
    }
}

// ---------------- k5: exclusive prefix across chunks ------------------------
__global__ void prefix_kernel(float* __restrict__ cse,
                              float* __restrict__ csx) {
    const int gid = blockIdx.x * 256 + threadIdx.x;   // 4096 = (b,d) pairs
    const int b = gid >> 9;
    const int d = gid & 511;
    float run = 0.f;
    for (int c = 0; c < 16; ++c) {
        const long long idx = ((long long)(b * 16 + c)) * D_DIM + d;
        const float v = csx[idx];
        csx[idx] = run;
        run += v;
    }
    if (blockIdx.x == 0 && threadIdx.x < B_DIM) {
        const int bb = threadIdx.x;
        float r = 0.f;
        for (int c = 0; c < 16; ++c) {
            const float v = cse[bb * 16 + c];
            cse[bb * 16 + c] = r;
            r += v;
        }
    }
}

// ---------------- k6: per-chunk scan -> context -----------------------------
__global__ __launch_bounds__(512)
void scan_kernel(const float* __restrict__ x,
                 const float* __restrict__ e_buf,
                 const float* __restrict__ cse,
                 const float* __restrict__ csx,
                 float* __restrict__ out) {
    __shared__ float es[CHUNK];
    const int b = blockIdx.x >> 4;
    const int c = blockIdx.x & 15;
    if (threadIdx.x < CHUNK)
        es[threadIdx.x] = e_buf[b * T_DIM + c * CHUNK + threadIdx.x];
    __syncthreads();
    const int d = threadIdx.x;                       // 512 threads == D
    float num = csx[(long long)blockIdx.x * D_DIM + d];
    float den = cse[blockIdx.x];
    const float* xp = x + ((long long)b * T_DIM + c * CHUNK) * D_DIM + d;
    float* op = out + ((long long)b * T_DIM + c * CHUNK) * D_DIM + d;
    for (int t = 0; t < CHUNK; ++t) {
        const float e = es[t];
        den += e;
        num = fmaf(e, xp[(long long)t * D_DIM], num);
        op[(long long)t * D_DIM] = num / den;
    }
}

// ---------------------------------------------------------------------------
extern "C" void kernel_launch(void* const* d_in, const int* in_sizes, int n_in,
                              void* d_out, int out_size, void* d_ws, size_t ws_size,
                              hipStream_t stream) {
    (void)in_sizes; (void)n_in; (void)out_size; (void)ws_size;
    const float* x  = (const float*)d_in[0];
    const float* W1 = (const float*)d_in[1];
    const float* b1 = (const float*)d_in[2];
    const float* w2 = (const float*)d_in[3];
    const float* b2 = (const float*)d_in[4];
    float* out = (float*)d_out;

    char* ws = (char*)d_ws;
    unsigned short* w1t  = (unsigned short*)(ws + 0);        // 512 KB bf16 W1^T
    float*          sbuf = (float*)(ws + 524288);            // 128 KB scores
    float*          mbuf = (float*)(ws + 655360);            // row max
    float*          ebuf = (float*)(ws + 655616);            // 128 KB e
    float*          cse  = (float*)(ws + 786688);            // chunk e sums
    float*          csx  = (float*)(ws + 787200);            // 256 KB chunk e*x

    hipLaunchKernelGGL(cvt_w1_kernel, dim3(512), dim3(256), 0, stream, W1, w1t);
    hipLaunchKernelGGL(scores_kernel, dim3(NTOK / 64), dim3(128), 131072, stream,
                       x, w1t, b1, w2, b2, sbuf);
    hipLaunchKernelGGL(rowmax_kernel, dim3(B_DIM), dim3(256), 0, stream, sbuf, mbuf);
    hipLaunchKernelGGL(chunksum_kernel, dim3(NCHUNK), dim3(256), 0, stream,
                       x, sbuf, mbuf, ebuf, cse, csx);
    hipLaunchKernelGGL(prefix_kernel, dim3(16), dim3(256), 0, stream, cse, csx);
    hipLaunchKernelGGL(scan_kernel, dim3(NCHUNK), dim3(512), 0, stream,
                       x, ebuf, cse, csx, out);
}
